// SAGE_model_86638080295291
// MI455X (gfx1250) — matrix operand order repaired and checked by reading, not compiled, not run on hardware
//
#include <hip/hip_runtime.h>

typedef __attribute__((ext_vector_type(16))) __bf16 v16bf;
typedef __attribute__((ext_vector_type(8)))  float  v8f;

// ---------------------------------------------------------------------------
// fp32 -> bf16 weight conversion
// ---------------------------------------------------------------------------
__global__ void cvt_bf16_kernel(const float* __restrict__ w, __bf16* __restrict__ o, int n) {
    int i = blockIdx.x * blockDim.x + threadIdx.x;
    if (i < n) o[i] = (__bf16)w[i];
}

// ---------------------------------------------------------------------------
// in-degree count + inverse (clamped to >= 1)
// ---------------------------------------------------------------------------
__global__ void deg_count_kernel(const int* __restrict__ dst, float* __restrict__ deg, int E) {
    int i = blockIdx.x * blockDim.x + threadIdx.x;
    if (i < E) atomicAdd(&deg[dst[i]], 1.0f);
}

__global__ void deg_inv_kernel(const float* __restrict__ deg, float* __restrict__ dinv, int n) {
    int i = blockIdx.x * blockDim.x + threadIdx.x;
    if (i < n) dinv[i] = 1.0f / fmaxf(deg[i], 1.0f);
}

// ---------------------------------------------------------------------------
// scatter-add: agg[dst] += h[src]  (one thread = one edge x 4 channels)
// ---------------------------------------------------------------------------
__global__ void scatter_add_kernel(const float* __restrict__ h, const int* __restrict__ src,
                                   const int* __restrict__ dst, float* __restrict__ agg,
                                   int E, int cpeSh) {
    int i = blockIdx.x * blockDim.x + threadIdx.x;
    int total = E << cpeSh;
    if (i >= total) return;
    int e = i >> cpeSh;
    int c = (i & ((1 << cpeSh) - 1)) << 2;
    int din = 1 << (cpeSh + 2);
    const float* hp = h   + (size_t)src[e] * din + c;
    float*       ap = agg + (size_t)dst[e] * din + c;
    float4 v = *(const float4*)hp;
    atomicAdd(ap + 0, v.x);
    atomicAdd(ap + 1, v.y);
    atomicAdd(ap + 2, v.z);
    atomicAdd(ap + 3, v.w);
}

// ---------------------------------------------------------------------------
// SAGE layer: out = relu( (agg*deg_inv) @ Wl^T + bl + h @ Wr^T )
// Block = 128 threads (4 waves) per 16-row node tile; waves split the output
// column tiles. bf16 WMMA, fp32 accumulate, two independent accumulators.
// A frag (16-bit 16x32): lane half hi -> per-VGPR K base = {2j | 16+2j} + 8*hi
//   -> two contiguous 16B runs in a row-major LDS tile  => 2x ds_load_b128
// B frag (32x16): col = lane&15, K runs = k0 + 8*hi + [0,8) and k0 + 16 + 8*hi + [0,8)
//   -> two 16B runs of the bf16 weight row              => 2x global_load_b128
// D: row = r + 8*hi, col = lane&15
// ---------------------------------------------------------------------------
__global__ __launch_bounds__(128) void sage_wmma_kernel(
    const float* __restrict__ h,     // N x din
    const float* __restrict__ agg,   // N x din (unscaled neighbor sums)
    const float* __restrict__ dinv,  // N
    const __bf16* __restrict__ Wl,   // dout x din (bf16, row-major)
    const __bf16* __restrict__ Wr,   // dout x din (bf16, row-major)
    const float* __restrict__ bias,  // dout
    float* __restrict__ out,         // N x dout
    int n, int din, int dinSh, int dout)
{
    __shared__ __align__(16) __bf16 sH[16 * 256];
    __shared__ __align__(16) __bf16 sA[16 * 256];
    const int tid  = threadIdx.x;
    const int wave = tid >> 5;
    const int lane = tid & 31;
    const int row0 = blockIdx.x << 4;

    // stage 16 rows of h and deg-scaled agg into LDS as bf16 (all 4 waves)
    for (int idx = tid; idx < (din << 4); idx += 128) {
        int r = idx >> dinSh;
        int k = idx & (din - 1);
        int row = row0 + r;
        float hv = 0.0f, av = 0.0f;
        if (row < n) {
            hv = h[(size_t)row * din + k];
            av = agg[(size_t)row * din + k] * dinv[row];
        }
        sH[(r << dinSh) + k] = (__bf16)hv;
        sA[(r << dinSh) + k] = (__bf16)av;
    }
    __syncthreads();

    const int hi = lane >> 4;
    const int m  = lane & 15;
    const unsigned int* sH32 = (const unsigned int*)sH;
    const unsigned int* sA32 = (const unsigned int*)sA;

    for (int n0 = wave << 4; n0 < dout; n0 += 64) {
        v8f accL = {};
        v8f accR = {};
        const int col = n0 + m;
        for (int k0 = 0; k0 < din; k0 += 32) {
            union { v16bf v; uint4 q[2]; } aA, aH, bL, bR;
            // A fragments: two 16B runs per lane from the LDS tile
            const int base = ((m << dinSh) + k0 + (hi << 3)) >> 1;   // dword idx, 16B aligned
            const uint4* pA = (const uint4*)(sA32 + base);
            const uint4* pH = (const uint4*)(sH32 + base);
            aA.q[0] = pA[0];           // K = k0+8hi   .. +7
            aA.q[1] = pA[2];           // K = k0+16+8hi.. +7
            aH.q[0] = pH[0];
            aH.q[1] = pH[2];
            // B fragments: two 16B runs of the weight row (same K order as A)
            const uint4* pL = (const uint4*)(Wl + (size_t)col * din + k0 + (hi << 3));
            const uint4* pR = (const uint4*)(Wr + (size_t)col * din + k0 + (hi << 3));
            bL.q[0] = pL[0];
            bL.q[1] = pL[2];
            bR.q[0] = pR[0];
            bR.q[1] = pR[2];
            accL = __builtin_amdgcn_wmma_f32_16x16x32_bf16(false, aA.v, false, bL.v,
                                                           (short)0, accL, false, false);
            accR = __builtin_amdgcn_wmma_f32_16x16x32_bf16(false, aH.v, false, bR.v,
                                                           (short)0, accR, false, false);
        }
        float b = bias[col];
#pragma unroll
        for (int r = 0; r < 8; ++r) {
            int row = row0 + r + (hi << 3);
            if (row < n) {
                float v = accL[r] + accR[r] + b;
                out[(size_t)row * dout + col] = v > 0.0f ? v : 0.0f;
            }
        }
    }
}

// ---------------------------------------------------------------------------
// head: softmax(h @ Wout^T + bout), 4 classes; one wave per node
// ---------------------------------------------------------------------------
__global__ __launch_bounds__(256) void head_softmax_kernel(
    const float* __restrict__ h, const float* __restrict__ Wout,
    const float* __restrict__ bout, float* __restrict__ out, int n, int dlast)
{
    int wid  = blockIdx.x * (blockDim.x >> 5) + (threadIdx.x >> 5);
    int lane = threadIdx.x & 31;
    if (wid >= n) return;
    float a0 = 0.f, a1 = 0.f, a2 = 0.f, a3 = 0.f;
    const float* hr = h + (size_t)wid * dlast;
    for (int k = lane; k < dlast; k += 32) {
        float hv = hr[k];
        a0 += hv * Wout[k];
        a1 += hv * Wout[dlast + k];
        a2 += hv * Wout[2 * dlast + k];
        a3 += hv * Wout[3 * dlast + k];
    }
#pragma unroll
    for (int off = 16; off > 0; off >>= 1) {
        a0 += __shfl_down(a0, off, 32);
        a1 += __shfl_down(a1, off, 32);
        a2 += __shfl_down(a2, off, 32);
        a3 += __shfl_down(a3, off, 32);
    }
    if (lane == 0) {
        a0 += bout[0]; a1 += bout[1]; a2 += bout[2]; a3 += bout[3];
        float mx = fmaxf(fmaxf(a0, a1), fmaxf(a2, a3));
        float e0 = __expf(a0 - mx), e1 = __expf(a1 - mx);
        float e2 = __expf(a2 - mx), e3 = __expf(a3 - mx);
        float s = 1.0f / (e0 + e1 + e2 + e3);
        float* o = out + (size_t)wid * 4;
        o[0] = e0 * s; o[1] = e1 * s; o[2] = e2 * s; o[3] = e3 * s;
    }
}

// ---------------------------------------------------------------------------
extern "C" void kernel_launch(void* const* d_in, const int* in_sizes, int n_in,
                              void* d_out, int out_size, void* d_ws, size_t ws_size,
                              hipStream_t stream)
{
    const int N = 50000;
    const float* x = (const float*)d_in[0];
    const float* Wl[5]; const float* bl[5]; const float* Wr[5];
    for (int i = 0; i < 5; ++i) {
        Wl[i] = (const float*)d_in[1 + 3 * i];
        bl[i] = (const float*)d_in[2 + 3 * i];
        Wr[i] = (const float*)d_in[3 + 3 * i];
    }
    const float* Wout = (const float*)d_in[16];
    const float* bout = (const float*)d_in[17];
    const int* edge = (const int*)d_in[18];
    const int E = in_sizes[18] / 2;
    const int* src = edge;       // edge_index[0]
    const int* dst = edge + E;   // edge_index[1]

    static const int DI[5] = {128, 128, 64, 128, 256};
    static const int DO[5] = {128, 64, 128, 256, 512};

    // workspace layout
    char* ws = (char*)d_ws;
    size_t off = 0;
    float* deg  = (float*)(ws + off); off += (size_t)N * 4;
    float* dinv = (float*)(ws + off); off += (size_t)N * 4;
    float* hA   = (float*)(ws + off); off += (size_t)N * 512 * 4;
    float* hB   = (float*)(ws + off); off += (size_t)N * 256 * 4;
    float* agg  = (float*)(ws + off); off += (size_t)N * 256 * 4;
    __bf16* wb  = (__bf16*)(ws + off);

    // convert weights to bf16
    __bf16* wlb[5]; __bf16* wrb[5];
    size_t woff = 0;
    for (int i = 0; i < 5; ++i) {
        int cnt = DI[i] * DO[i];
        wlb[i] = wb + woff; woff += cnt;
        wrb[i] = wb + woff; woff += cnt;
        cvt_bf16_kernel<<<(cnt + 255) / 256, 256, 0, stream>>>(Wl[i], wlb[i], cnt);
        cvt_bf16_kernel<<<(cnt + 255) / 256, 256, 0, stream>>>(Wr[i], wrb[i], cnt);
    }

    // degrees
    hipMemsetAsync(deg, 0, (size_t)N * 4, stream);
    deg_count_kernel<<<(E + 255) / 256, 256, 0, stream>>>(dst, deg, E);
    deg_inv_kernel<<<(N + 255) / 256, 256, 0, stream>>>(deg, dinv, N);

    // layers (ping-pong: x -> hA -> hB -> hA -> hB -> hA)
    const float* hin = x;
    float* bufs[5] = { hA, hB, hA, hB, hA };
    for (int i = 0; i < 5; ++i) {
        int din = DI[i], dout = DO[i];
        int dinSh = (din == 64) ? 6 : (din == 128) ? 7 : 8;
        hipMemsetAsync(agg, 0, (size_t)N * din * 4, stream);
        int cpeSh = dinSh - 2;
        int total = E << cpeSh;
        scatter_add_kernel<<<(total + 255) / 256, 256, 0, stream>>>(hin, src, dst, agg, E, cpeSh);
        sage_wmma_kernel<<<(N + 15) / 16, 128, 0, stream>>>(
            hin, agg, dinv, wlb[i], wrb[i], bl[i], bufs[i], N, din, dinSh, dout);
        hin = bufs[i];
    }

    // head + softmax
    head_softmax_kernel<<<(N + 7) / 8, 256, 0, stream>>>(hin, Wout, bout, (float*)d_out, N, 512);
}
